// MyAttention_5746666242270
// MI455X (gfx1250) — compile-verified, matrix operand order checked by reading, not run on hardware
//
#include <hip/hip_runtime.h>
#include <hip/hip_bf16.h>

typedef unsigned short u16;
typedef __attribute__((ext_vector_type(16))) __bf16 v16bf;
typedef __attribute__((ext_vector_type(8)))  float  v8f;
typedef __attribute__((__vector_size__(16))) int i32x4;

#define Bsz  4
#define Ssz  2048
#define DIN  1024
#define Hh   16
#define Dd   64
#define Mrows (Bsz * Ssz)        // 8192

// Async LDS copy path (gfx1250: GLOBAL_LOAD_ASYNC_TO_LDS_B128, ASYNCcnt)
#if defined(__gfx1250__) && __has_builtin(__builtin_amdgcn_global_load_async_to_lds_b128)
#define USE_ASYNC 1
#else
#define USE_ASYNC 0
#endif

// ---------- helpers ----------
__device__ __forceinline__ u16 f2bf(float f) {
  unsigned u = __builtin_bit_cast(unsigned, f);
  u += 0x7FFFu + ((u >> 16) & 1u);           // round-to-nearest-even
  return (u16)(u >> 16);
}

// Load one WMMA A/B operand half-pair: two contiguous 16B chunks at k0 and k0+16.
// rowptr points to the bf16 row, k0 already includes 8*(lane>>4).
__device__ __forceinline__ v16bf load_ab(const u16* rowptr, int k0) {
  union { v16bf v; uint4 q[2]; } u;
  u.q[0] = *reinterpret_cast<const uint4*>(rowptr + k0);
  u.q[1] = *reinterpret_cast<const uint4*>(rowptr + k0 + 16);
  return u.v;
}

__device__ __forceinline__ v8f wmma_bf16(v16bf a, v16bf b, v8f c) {
  return __builtin_amdgcn_wmma_f32_16x16x32_bf16(false, a, false, b, (short)0, c, false, false);
}

__device__ __forceinline__ v8f zero8() {
  v8f z;
#pragma unroll
  for (int i = 0; i < 8; ++i) z[i] = 0.0f;
  return z;
}

// 16-byte global -> LDS copy: async on gfx1250 (writes LDS directly, ASYNCcnt),
// synchronous load+store fallback otherwise.
__device__ __forceinline__ void cp16_g2l(u16* ldst, const u16* gsrc) {
#if USE_ASYNC
  __builtin_amdgcn_global_load_async_to_lds_b128(
      (__attribute__((address_space(1))) i32x4*)(gsrc),
      (__attribute__((address_space(3))) i32x4*)(ldst),
      0, 0);
#else
  *reinterpret_cast<uint4*>(ldst) = *reinterpret_cast<const uint4*>(gsrc);
#endif
}

// Wait for outstanding async copies (this wave) then block barrier.
__device__ __forceinline__ void publish_stage() {
#if USE_ASYNC
#if __has_builtin(__builtin_amdgcn_s_wait_asynccnt)
  __builtin_amdgcn_s_wait_asynccnt(0);
#else
  asm volatile("s_wait_asynccnt 0x0" ::: "memory");
#endif
#endif
  __syncthreads();
}

// ---------- conversion kernels ----------
__global__ void cvt_bf16(const float* __restrict__ in, u16* __restrict__ out, int n) {
  int i = blockIdx.x * 256 + threadIdx.x;
  if (i < n) out[i] = f2bf(in[i]);
}

// w: [K,N] fp32 row-major -> wt: [N,K] bf16 row-major (transposed)
__global__ void cvt_wt(const float* __restrict__ w, u16* __restrict__ wt, int K, int N) {
  int i = blockIdx.x * 256 + threadIdx.x;
  if (i < K * N) {
    int kk = i / N, n = i % N;
    wt[(size_t)n * K + kk] = f2bf(w[i]);
  }
}

// ---------- QKV projection: one wave computes a 16x64 tile of X @ W ----------
// Xb: [8192,1024] bf16.  Wt: [1024,1024] bf16 (W transposed).
// Q,K stored [B,H,S,D]; V stored transposed [B,H,D,S].
__global__ __launch_bounds__(32) void qkv_gemm(
    const u16* __restrict__ Xb,
    const u16* __restrict__ Wqt, const u16* __restrict__ Wkt, const u16* __restrict__ Wvt,
    const float* __restrict__ bq, const float* __restrict__ bk, const float* __restrict__ bv,
    u16* __restrict__ Qo, u16* __restrict__ Ko, u16* __restrict__ Vto)
{
  const int lane = threadIdx.x & 31;
  const int lm = lane & 15, lh = lane >> 4;
  const int rowTile = blockIdx.x & 511;       // 512 row tiles of 16
  const int colTile = blockIdx.x >> 9;        // 16 col tiles of 64
  const int w = blockIdx.y;                   // 0=Q 1=K 2=V

  const u16*  Wt   = (w == 0) ? Wqt : (w == 1) ? Wkt : Wvt;
  const float* bia = (w == 0) ? bq  : (w == 1) ? bk  : bv;
  u16*        out  = (w == 0) ? Qo  : (w == 1) ? Ko  : Vto;

  const int m0 = rowTile * 16;
  const int c0 = colTile * 64;

  v8f acc[4];
#pragma unroll
  for (int t = 0; t < 4; ++t) acc[t] = zero8();

  const u16* arow = Xb + (size_t)(m0 + lm) * DIN;
  for (int kk = 0; kk < DIN; kk += 32) {
    v16bf a = load_ab(arow, kk + 8 * lh);
#pragma unroll
    for (int t = 0; t < 4; ++t) {
      const u16* brow = Wt + (size_t)(c0 + t * 16 + lm) * DIN;
      v16bf b = load_ab(brow, kk + 8 * lh);
      acc[t] = wmma_bf16(a, b, acc[t]);
    }
  }

#pragma unroll
  for (int t = 0; t < 4; ++t) {
    const int c = c0 + t * 16 + lm;           // 0..1023
    const float bval = bia[c];
    const int h = c >> 6, d = c & 63;
#pragma unroll
    for (int r = 0; r < 8; ++r) {
      const int row = m0 + r + 8 * lh;        // 0..8191
      const int bb = row >> 11, s = row & 2047;
      const u16 o = f2bf(acc[t][r] + bval);
      if (w == 2)
        out[((size_t)(bb * Hh + h) * Dd + d) * Ssz + s] = o;      // [B,H,D,S]
      else
        out[((size_t)(bb * Hh + h) * Ssz + s) * Dd + d] = o;      // [B,H,S,D]
    }
  }
}

// ---------- Flash attention: 4 waves/block, each wave owns 16 q-rows ----------
// Q,K: [B,H,S,D] bf16;  Vt: [B,H,D,S] bf16;  Ob: [B,S,H*D] bf16.
// Double-buffered async staging of 32-key blocks (K tile 32x64, Vt tile 64x32).
__global__ __launch_bounds__(128) void flash_attn(
    const u16* __restrict__ Q, const u16* __restrict__ Kb,
    const u16* __restrict__ Vt, u16* __restrict__ Ob)
{
  __shared__ u16 Klds[2][32 * 64];   // 32 keys x 64 d, double-buffered
  __shared__ u16 Vlds[2][64 * 32];   // 64 d x 32 keys (transposed), double-buffered
  __shared__ u16 Plds[4][16 * 32];   // per-wave P transpose scratch

  const int tid  = threadIdx.x;
  const int wave = tid >> 5;
  const int lane = tid & 31;
  const int lm = lane & 15, lh = lane >> 4;

  const int bh = blockIdx.x >> 5;              // 0..63   (b*16+h)
  const int qb = blockIdx.x & 31;              // 32 q-blocks of 64 rows
  const int q0 = qb * 64 + wave * 16;
  const size_t bhS = (size_t)bh * Ssz;

  // Q tile held in A-layout registers for the whole kernel (K-dim = 64 -> two halves)
  const u16* qrow = Q + (bhS + q0 + lm) * Dd;
  const v16bf qa0 = load_ab(qrow, 0  + 8 * lh);
  const v16bf qa1 = load_ab(qrow, 32 + 8 * lh);

  v8f acc[4];
#pragma unroll
  for (int t = 0; t < 4; ++t) acc[t] = zero8();
  float mrow[8], lrow[8];
#pragma unroll
  for (int r = 0; r < 8; ++r) { mrow[r] = -__builtin_inff(); lrow[r] = 0.0f; }

  // stage one 32-key block into buffer `buf`: 512 x 16B chunks over 128 threads
  auto stage = [&](int buf, int ks0) {
#pragma unroll
    for (int it = 0; it < 4; ++it) {
      const int i = tid + it * 128;
      if (i < 256) {
        const int rr = i >> 3, ch = i & 7;
        cp16_g2l(&Klds[buf][rr * 64 + ch * 8],
                 Kb + (bhS + ks0 + rr) * Dd + ch * 8);
      } else {
        const int j = i - 256;
        const int rr = j >> 2, ch = j & 3;
        cp16_g2l(&Vlds[buf][rr * 32 + ch * 8],
                 Vt + ((size_t)bh * Dd + rr) * Ssz + ks0 + ch * 8);
      }
    }
  };

  constexpr int NB = Ssz / 32;        // 64 key blocks
  stage(0, 0);
  for (int kbk = 0; kbk < NB; ++kbk) {
    const int cur = kbk & 1;
    publish_stage();                  // buffer `cur` resident; `cur^1` free
    if (kbk + 1 < NB) stage(cur ^ 1, (kbk + 1) * 32);

    // scores S = Q(16x64) @ K_blk^T(64x32): two 16x16 C tiles
    v8f s0 = zero8(), s1 = zero8();
    s0 = wmma_bf16(qa0, load_ab(&Klds[cur][(lm) * 64],      0  + 8 * lh), s0);
    s0 = wmma_bf16(qa1, load_ab(&Klds[cur][(lm) * 64],      32 + 8 * lh), s0);
    s1 = wmma_bf16(qa0, load_ab(&Klds[cur][(16 + lm) * 64], 0  + 8 * lh), s1);
    s1 = wmma_bf16(qa1, load_ab(&Klds[cur][(16 + lm) * 64], 32 + 8 * lh), s1);

    // online softmax; row m = r + 8*lh lives in the 16 lanes of this half
#pragma unroll
    for (int r = 0; r < 8; ++r) {
      float a0 = s0[r] * 0.125f;   // 1/sqrt(64)
      float a1 = s1[r] * 0.125f;
      float mx = fmaxf(a0, a1);
      mx = fmaxf(mx, __shfl_xor(mx, 1));
      mx = fmaxf(mx, __shfl_xor(mx, 2));
      mx = fmaxf(mx, __shfl_xor(mx, 4));
      mx = fmaxf(mx, __shfl_xor(mx, 8));
      const float mnew  = fmaxf(mrow[r], mx);
      const float alpha = __expf(mrow[r] - mnew);
      mrow[r] = mnew;
      const float p0 = __expf(a0 - mnew);
      const float p1 = __expf(a1 - mnew);
      s0[r] = p0; s1[r] = p1;
      float rs = p0 + p1;
      rs += __shfl_xor(rs, 1);
      rs += __shfl_xor(rs, 2);
      rs += __shfl_xor(rs, 4);
      rs += __shfl_xor(rs, 8);
      lrow[r] = lrow[r] * alpha + rs;
      acc[0][r] *= alpha; acc[1][r] *= alpha; acc[2][r] *= alpha; acc[3][r] *= alpha;
    }

    // transpose P (C-layout f32) -> A-layout bf16 via per-wave LDS patch
    u16* P = &Plds[wave][0];
#pragma unroll
    for (int r = 0; r < 8; ++r) {
      const int m = r + 8 * lh;
      P[m * 32 + lm]      = f2bf(s0[r]);
      P[m * 32 + 16 + lm] = f2bf(s1[r]);
    }
    const v16bf pa = load_ab(&Plds[wave][lm * 32], 8 * lh);

    // O += P(16x32) @ V_blk(32x64): 4 n-tiles, K=32 in one wmma each
#pragma unroll
    for (int t = 0; t < 4; ++t) {
      v16bf bv = load_ab(&Vlds[cur][(t * 16 + lm) * 32], 8 * lh);
      acc[t] = wmma_bf16(pa, bv, acc[t]);
    }
  }

  // epilogue: O/l -> Ob[b, s, h*64+d] bf16
  const int bb = bh >> 4, h = bh & 15;
#pragma unroll
  for (int t = 0; t < 4; ++t) {
    const int col = h * 64 + t * 16 + lm;
#pragma unroll
    for (int r = 0; r < 8; ++r) {
      const int s = q0 + r + 8 * lh;
      Ob[((size_t)bb * Ssz + s) * (Hh * Dd) + col] = f2bf(acc[t][r] / lrow[r]);
    }
  }
}

// ---------- output projection: [8192,1024] @ [1024,64] + bo -> f32 ----------
__global__ __launch_bounds__(32) void out_gemm(
    const u16* __restrict__ Ob, const u16* __restrict__ Wot,
    const float* __restrict__ bo, float* __restrict__ out)
{
  const int lane = threadIdx.x & 31;
  const int lm = lane & 15, lh = lane >> 4;
  const int m0 = blockIdx.x * 16;

  v8f acc[4];
#pragma unroll
  for (int t = 0; t < 4; ++t) acc[t] = zero8();

  const u16* arow = Ob + (size_t)(m0 + lm) * (Hh * Dd);
  for (int kk = 0; kk < Hh * Dd; kk += 32) {
    v16bf a = load_ab(arow, kk + 8 * lh);
#pragma unroll
    for (int t = 0; t < 4; ++t) {
      v16bf b = load_ab(Wot + (size_t)(t * 16 + lm) * (Hh * Dd), kk + 8 * lh);
      acc[t] = wmma_bf16(a, b, acc[t]);
    }
  }
#pragma unroll
  for (int t = 0; t < 4; ++t) {
    const int c = t * 16 + lm;
    const float bb = bo[c];
#pragma unroll
    for (int r = 0; r < 8; ++r) {
      const int row = m0 + r + 8 * lh;
      out[(size_t)row * Dd + c] = acc[t][r] + bb;
    }
  }
}

// ---------- host ----------
extern "C" void kernel_launch(void* const* d_in, const int* in_sizes, int n_in,
                              void* d_out, int out_size, void* d_ws, size_t ws_size,
                              hipStream_t stream) {
  (void)in_sizes; (void)n_in; (void)out_size; (void)ws_size;
  const float* x  = (const float*)d_in[0];
  const float* wq = (const float*)d_in[1];
  const float* bq = (const float*)d_in[2];
  const float* wk = (const float*)d_in[3];
  const float* bk = (const float*)d_in[4];
  const float* wv = (const float*)d_in[5];
  const float* bv = (const float*)d_in[6];
  const float* wo = (const float*)d_in[7];
  const float* bo = (const float*)d_in[8];
  float* out = (float*)d_out;

  char* ws = (char*)d_ws;
  const size_t szXb = (size_t)Mrows * DIN * 2;            // 16 MB
  const size_t szW  = (size_t)DIN * (Hh * Dd) * 2;        // 2 MB each
  const size_t szWo = (size_t)(Hh * Dd) * Dd * 2;         // 128 KB (stored [64,1024])
  const size_t szT  = (size_t)Bsz * Hh * Ssz * Dd * 2;    // 16 MB each

  u16* Xb  = (u16*)(ws);
  u16* Wqt = (u16*)(ws + szXb);
  u16* Wkt = (u16*)(ws + szXb + szW);
  u16* Wvt = (u16*)(ws + szXb + 2 * szW);
  u16* Wot = (u16*)(ws + szXb + 3 * szW);
  u16* Qd  = (u16*)(ws + szXb + 3 * szW + szWo);
  u16* Kd  = (u16*)(ws + szXb + 3 * szW + szWo + szT);
  u16* Vtd = (u16*)(ws + szXb + 3 * szW + szWo + 2 * szT);
  u16* Obd = (u16*)(ws + szXb + 3 * szW + szWo + 3 * szT);

  const int nX = Mrows * DIN;                 // 8388608
  cvt_bf16<<<(nX + 255) / 256, 256, 0, stream>>>(x, Xb, nX);
  cvt_wt<<<(DIN * Hh * Dd + 255) / 256, 256, 0, stream>>>(wq, Wqt, DIN, Hh * Dd);
  cvt_wt<<<(DIN * Hh * Dd + 255) / 256, 256, 0, stream>>>(wk, Wkt, DIN, Hh * Dd);
  cvt_wt<<<(DIN * Hh * Dd + 255) / 256, 256, 0, stream>>>(wv, Wvt, DIN, Hh * Dd);
  cvt_wt<<<(Hh * Dd * Dd + 255) / 256, 256, 0, stream>>>(wo, Wot, Hh * Dd, Dd);

  qkv_gemm<<<dim3(512 * 16, 3), 32, 0, stream>>>(Xb, Wqt, Wkt, Wvt, bq, bk, bv, Qd, Kd, Vtd);
  flash_attn<<<(Bsz * Hh) * (Ssz / 64), 128, 0, stream>>>(Qd, Kd, Vtd, Obd);
  out_gemm<<<Mrows / 16, 32, 0, stream>>>(Obd, Wot, bo, out);
}
